// GCN_37220186587698
// MI455X (gfx1250) — matrix operand order repaired and checked by reading, not compiled, not run on hardware
//
#include <hip/hip_runtime.h>
#include <hip/hip_fp16.h>

// GCN forward for MI455X (gfx1250, wave32).
// out = softmax( A @ ((relu(A @ ((H W1) scaled))) W2 scaled) ), A = D^-1/2 (adj+I) D^-1/2
// Raw-adj matmuls (entries {0,1} exact in f16) on v_wmma_f32_16x16x32_f16, f32 accum.
// Software-pipelined (double-buffered) K loop + split-K (2 halves) for 1024-wave parallelism.

#define NN    8192
#define FIN   64
#define FHID  64
#define FOUT  16

typedef __attribute__((ext_vector_type(16))) _Float16 v16h;
typedef __attribute__((ext_vector_type(8)))  float    v8f;
typedef __attribute__((ext_vector_type(2)))  __fp16   v2h;   // cvt_pkrtz return type

union AFrag { v16h v; v2h  p[8]; };   // 16x32 f16 A fragment (8 VGPRs)
union BFrag { v16h v; uint4 u[2]; };  // 32x16 f16 B fragment (8 VGPRs)
struct ARaw { float4 a0, a1, a2, a3; };

// ---------------- degree pass: dinv[i] = rsqrt(1 + sum_j adj[i][j]) ----------------
__global__ __launch_bounds__(256)
void k_deg(const float* __restrict__ adj, float* __restrict__ dinv) {
    __shared__ float red[256];
    const int row = blockIdx.x;
    const float4* r = reinterpret_cast<const float4*>(adj + (size_t)row * NN);
    float s = 0.f;
    for (int i = threadIdx.x; i < NN / 4; i += 256) {
        float4 v = r[i];
        s += v.x + v.y + v.z + v.w;
    }
    red[threadIdx.x] = s;
    __syncthreads();
    for (int off = 128; off > 0; off >>= 1) {
        if (threadIdx.x < off) red[threadIdx.x] += red[threadIdx.x + off];
        __syncthreads();
    }
    if (threadIdx.x == 0) dinv[row] = rsqrtf(red[0] + 1.0f);
}

// ---------------- Xs1T[c][j] = f16( dinv[j] * (H @ W1)[j][c] ), full f32 dot ----------------
__global__ __launch_bounds__(256)
void k_xw1(const float* __restrict__ H, const float* __restrict__ W1,
           const float* __restrict__ dinv, _Float16* __restrict__ Xs1T) {
    __shared__ float w1s[FIN * FHID];   // 16 KB
    __shared__ float hs[4][FIN];        // 1 KB
    const int t = threadIdx.x;
    for (int i = t; i < FIN * FHID; i += 256) w1s[i] = W1[i];
    const int row0 = blockIdx.x * 4;
    for (int i = t; i < 4 * FIN; i += 256)
        hs[i / FIN][i % FIN] = H[(size_t)(row0 + i / FIN) * FIN + (i % FIN)];
    __syncthreads();
    const int rl  = t >> 6;
    const int c   = t & 63;
    const int row = row0 + rl;
    float acc = 0.f;
#pragma unroll
    for (int k = 0; k < FIN; ++k) acc += hs[rl][k] * w1s[k * FHID + c];
    Xs1T[(size_t)c * NN + row] = (_Float16)(acc * dinv[row]);
}

// Raw fp32 loads for one 16x32 A tile (per-lane: 4x float4).
__device__ __forceinline__ ARaw load_a_raw(const float* __restrict__ arow, int kb) {
    const float4* ap  = reinterpret_cast<const float4*>(arow + kb);
    const float4* ap2 = reinterpret_cast<const float4*>(arow + kb + 16);
    ARaw r; r.a0 = ap[0]; r.a1 = ap[1]; r.a2 = ap2[0]; r.a3 = ap2[1];
    return r;
}

// Pack into the ISA-layout 16x32 f16 A fragment.
// lanes 0-15: M=lane, K = {0..7, 16..23}; lanes 16-31: M=lane-16, K = {8..15, 24..31}.
__device__ __forceinline__ AFrag cvt_a(const ARaw& r) {
    AFrag af;
    af.p[0] = __builtin_amdgcn_cvt_pkrtz(r.a0.x, r.a0.y);
    af.p[1] = __builtin_amdgcn_cvt_pkrtz(r.a0.z, r.a0.w);
    af.p[2] = __builtin_amdgcn_cvt_pkrtz(r.a1.x, r.a1.y);
    af.p[3] = __builtin_amdgcn_cvt_pkrtz(r.a1.z, r.a1.w);
    af.p[4] = __builtin_amdgcn_cvt_pkrtz(r.a2.x, r.a2.y);
    af.p[5] = __builtin_amdgcn_cvt_pkrtz(r.a2.z, r.a2.w);
    af.p[6] = __builtin_amdgcn_cvt_pkrtz(r.a3.x, r.a3.y);
    af.p[7] = __builtin_amdgcn_cvt_pkrtz(r.a3.z, r.a3.w);
    return af;
}

__device__ __forceinline__ BFrag load_b(const _Float16* __restrict__ base, int kb) {
    const uint4* bp = reinterpret_cast<const uint4*>(base + kb);
    BFrag bf; bf.u[0] = bp[0]; bf.u[1] = bp[2];
    return bf;
}

// ---------------- layer 1: Y = adj @ Xs1 ; H1 = relu(d*(Y + Xs1)) ; Xs2 = d*(H1 @ W2) ----------------
// block = 256 threads (8 waves): 4 row tiles x 2 K-halves. Each wave: 16 rows x 64 cols.
__global__ __launch_bounds__(256)
void k_mm1(const float* __restrict__ adj, const _Float16* __restrict__ Xs1T,
           const float* __restrict__ dinv, const float* __restrict__ W2,
           _Float16* __restrict__ Xs2T) {
    __shared__ float redb[64][64];      // 16 KB: K-half-1 partial sums
    __shared__ float h1s[64][64];       // 16 KB: block's H1 tile
    __shared__ float w2s[FHID * FOUT];  // 4 KB
    const int t = threadIdx.x;
    for (int i = t; i < FHID * FOUT; i += 256) w2s[i] = W2[i];

    const int lane    = t & 31;
    const int wave    = t >> 5;        // 0..7
    const int rtile   = wave & 3;      // row tile in block
    const int khalf   = wave >> 2;     // K half
    const int rowBase = blockIdx.x * 64 + rtile * 16;
    const int mrow    = rowBase + (lane & 15);
    const int hiHalf  = lane >> 4;
    const float* arow = adj + (size_t)mrow * NN + hiHalf * 8;
    const int kcol    = hiHalf ? 8 : 0;

    const _Float16* bbase[4];
#pragma unroll
    for (int n = 0; n < 4; ++n)
        bbase[n] = Xs1T + (size_t)(n * 16 + (lane & 15)) * NN + kcol;

    const int k0 = khalf * (NN / 2);
    const int k1 = k0 + (NN / 2);

    v8f acc[4] = {};
    // software pipeline: stage next tile's loads while current tile computes
    ARaw  ar = load_a_raw(arow, k0);
    BFrag bf[4];
#pragma unroll
    for (int n = 0; n < 4; ++n) bf[n] = load_b(bbase[n], k0);

    for (int kb = k0; kb < k1 - 32; kb += 32) {
        ARaw arn = load_a_raw(arow, kb + 32);
        BFrag bfn[4];
#pragma unroll
        for (int n = 0; n < 4; ++n) bfn[n] = load_b(bbase[n], kb + 32);
        __builtin_prefetch((const void*)(arow + kb + 96), 0, 3);  // speculative, non-faulting

        AFrag af = cvt_a(ar);
#pragma unroll
        for (int n = 0; n < 4; ++n)
            acc[n] = __builtin_amdgcn_wmma_f32_16x16x32_f16(
                false, af.v, false, bf[n].v, (short)0, acc[n], false, false);

        ar = arn;
#pragma unroll
        for (int n = 0; n < 4; ++n) bf[n] = bfn[n];
    }
    {   // peeled last tile
        AFrag af = cvt_a(ar);
#pragma unroll
        for (int n = 0; n < 4; ++n)
            acc[n] = __builtin_amdgcn_wmma_f32_16x16x32_f16(
                false, af.v, false, bf[n].v, (short)0, acc[n], false, false);
    }

    // split-K reduction: khalf=1 waves deposit partials in LDS
    if (khalf) {
#pragma unroll
        for (int n = 0; n < 4; ++n) {
            const int col = n * 16 + (lane & 15);
#pragma unroll
            for (int v = 0; v < 8; ++v)
                redb[rtile * 16 + v + hiHalf * 8][col] = acc[n][v];
        }
    }
    __syncthreads();

    // epilogue (khalf=0 waves): combine halves, identity term, scale, relu -> h1s
    if (!khalf) {
#pragma unroll
        for (int n = 0; n < 4; ++n) {
            const int col = n * 16 + (lane & 15);
#pragma unroll
            for (int v = 0; v < 8; ++v) {
                const int rl  = rtile * 16 + v + hiHalf * 8;
                const int row = rowBase + v + hiHalf * 8;
                float y  = acc[n][v] + redb[rl][col] + (float)Xs1T[(size_t)col * NN + row];
                float h1 = dinv[row] * y;
                h1s[rl][col] = h1 > 0.f ? h1 : 0.f;
            }
        }
    }
    __syncthreads();

    // fused layer-2 input: Xs2T[c][row] = f16( dinv[row] * (H1 @ W2)[row][c] )
    for (int idx = t; idx < 64 * FOUT; idx += 256) {
        const int rl = idx >> 4;
        const int c  = idx & 15;
        float s = 0.f;
#pragma unroll
        for (int k = 0; k < FHID; ++k) s += h1s[rl][k] * w2s[k * FOUT + c];
        const int grow = blockIdx.x * 64 + rl;
        Xs2T[(size_t)c * NN + grow] = (_Float16)(dinv[grow] * s);
    }
}

// ---------------- layer 2: H2 = d*((adj @ Xs2) + Xs2) ; out = softmax(H2) ----------------
// block = 256 threads (8 waves): 4 row tiles x 2 K-halves. Each wave: 16 rows x 16 cols.
__global__ __launch_bounds__(256)
void k_mm2(const float* __restrict__ adj, const _Float16* __restrict__ Xs2T,
           const float* __restrict__ dinv, float* __restrict__ out) {
    __shared__ float redb2[64][16];     // 4 KB: K-half-1 partials
    const int t       = threadIdx.x;
    const int lane    = t & 31;
    const int wave    = t >> 5;
    const int rtile   = wave & 3;
    const int khalf   = wave >> 2;
    const int rowBase = blockIdx.x * 64 + rtile * 16;
    const int mrow    = rowBase + (lane & 15);
    const int hiHalf  = lane >> 4;
    const float* arow = adj + (size_t)mrow * NN + hiHalf * 8;
    const int col     = lane & 15;
    const _Float16* bcol = Xs2T + (size_t)col * NN + (hiHalf ? 8 : 0);

    const int k0 = khalf * (NN / 2);
    const int k1 = k0 + (NN / 2);

    v8f acc = {};
    ARaw  ar = load_a_raw(arow, k0);
    BFrag bf = load_b(bcol, k0);

    for (int kb = k0; kb < k1 - 32; kb += 32) {
        ARaw  arn = load_a_raw(arow, kb + 32);
        BFrag bfn = load_b(bcol, kb + 32);
        __builtin_prefetch((const void*)(arow + kb + 96), 0, 3);

        AFrag af = cvt_a(ar);
        acc = __builtin_amdgcn_wmma_f32_16x16x32_f16(
            false, af.v, false, bf.v, (short)0, acc, false, false);
        ar = arn; bf = bfn;
    }
    {
        AFrag af = cvt_a(ar);
        acc = __builtin_amdgcn_wmma_f32_16x16x32_f16(
            false, af.v, false, bf.v, (short)0, acc, false, false);
    }

    if (khalf) {
#pragma unroll
        for (int v = 0; v < 8; ++v)
            redb2[rtile * 16 + v + hiHalf * 8][col] = acc[v];
    }
    __syncthreads();

    if (!khalf) {
        // softmax over 16 classes: each 16-lane group holds one row of C
#pragma unroll
        for (int v = 0; v < 8; ++v) {
            const int rl  = rtile * 16 + v + hiHalf * 8;
            const int row = rowBase + v + hiHalf * 8;
            float h2 = dinv[row] * (acc[v] + redb2[rl][col] +
                                    (float)Xs2T[(size_t)col * NN + row]);
            float m = h2;
#pragma unroll
            for (int off = 1; off < 16; off <<= 1) m = fmaxf(m, __shfl_xor(m, off, 32));
            float e = __expf(h2 - m);
            float s = e;
#pragma unroll
            for (int off = 1; off < 16; off <<= 1) s += __shfl_xor(s, off, 32);
            out[(size_t)row * FOUT + col] = e / s;
        }
    }
}

extern "C" void kernel_launch(void* const* d_in, const int* in_sizes, int n_in,
                              void* d_out, int out_size, void* d_ws, size_t ws_size,
                              hipStream_t stream) {
    const float* H   = (const float*)d_in[0];   // [8192, 64]
    const float* adj = (const float*)d_in[1];   // [8192, 8192]
    const float* W1  = (const float*)d_in[2];   // [64, 64]
    const float* W2  = (const float*)d_in[3];   // [64, 16]
    float* out = (float*)d_out;                 // [8192, 16]

    char* ws = (char*)d_ws;
    float*     dinv = (float*)ws;                                   // 32 KB
    _Float16*  Xs1T = (_Float16*)(ws + 32 * 1024);                  // 64 x 8192 f16, 1 MB
    _Float16*  Xs2T = (_Float16*)(ws + 32 * 1024 + NN * FHID * 2);  // 16 x 8192 f16, 256 KB

    k_deg <<<NN,      256, 0, stream>>>(adj, dinv);
    k_xw1 <<<NN / 4,  256, 0, stream>>>(H, W1, dinv, Xs1T);
    k_mm1 <<<NN / 64, 256, 0, stream>>>(adj, Xs1T, dinv, W2, Xs2T);
    k_mm2 <<<NN / 64, 256, 0, stream>>>(adj, Xs2T, dinv, out);
}
